// FullyConnectedTensorProduct_80736795230260
// MI455X (gfx1250) — matrix operand order repaired and checked by reading, not compiled, not run on hardware
//
#include <hip/hip_runtime.h>
#include <hip/hip_bf16.h>

typedef __attribute__((ext_vector_type(16))) __bf16 v16bf;
typedef __attribute__((ext_vector_type(8)))  float  v8f;

#define NPATH 13
// paths in reference INS order: (l1,l2,lo); io == lo for this irrep set
constexpr int cL1[NPATH]    = {0,0,0,1,1,1,1,1,2,2,2,2,2};
constexpr int cL2[NPATH]    = {0,1,2,0,1,1,2,3,0,1,2,2,3};
constexpr int cLO[NPATH]    = {0,1,2,1,0,2,1,2,2,1,0,2,1};
constexpr int cSLOT[NPATH]  = {0,0,0,1,1,1,2,2,3,3,2,4,4};                 // slot within output irrep
constexpr int cGOFF[NPATH]  = {0,1,4,9,18,21,36,45,60,85,100,105,130};     // per-node G offsets (145 floats)
constexpr int cCGOFF[NPATH] = {0,1,10,35,44,53,98,143,248,273,318,343,468};// cg table offsets (573 floats)
constexpr int cXL[3]   = {0,1,4};        // offset into per-(z,u) 9-float x vector
constexpr int cYOFF[4] = {0,1,4,9};
constexpr int cOUT[3]  = {0,128,512};
constexpr int cIOP[3][5] = {{0,4,10,0,0},{1,3,6,9,12},{2,5,7,8,11}};       // io -> path per slot
constexpr int cWFB[3]  = {0,96,256};     // weight-fragment base (frag units) per io

#define G_FLOATS   (16*145)
#define T_BYTE_OFF (G_FLOATS*4)             // 9280, 32B aligned
#define N_TFRAG    172                      // io0:12, io1:3*20, io2:5*20
#define SMEM_BYTES (T_BYTE_OFF + N_TFRAG*1024)   // 185408 < 320KB/WGP
#define N_WFRAG    416                      // (12+20+20)*8
#define WF_BYTE_OFF 4096

// ---------------- device Wigner-3j (Racah + real-basis transform) ----------------
static __device__ int imax2(int a,int b){return a>b?a:b;}
static __device__ int imin2(int a,int b){return a<b?a:b;}
static __device__ double dfact(int n){ double r=1.0; for(int i=2;i<=n;++i) r*=(double)i; return r; }

static __device__ double su2_cg(int j1,int m1,int j2,int m2,int j3,int m3){
  if (m3 != m1+m2) return 0.0;
  int vmin = imax2(imax2(-j1+j2+m3, -j1+m1), 0);
  int vmax = imin2(imin2(j2+j3+m1, j3-j1+j2), j3+m3);
  double C = sqrt((2.0*j3+1.0)*dfact(j3+j1-j2)*dfact(j3-j1+j2)*dfact(j1+j2-j3)
                  *dfact(j3+m3)*dfact(j3-m3)
                  /(dfact(j1+j2+j3+1)*dfact(j1-m1)*dfact(j1+m1)*dfact(j2-m2)*dfact(j2+m2)));
  double S = 0.0;
  #pragma clang loop unroll(disable)
  for (int v=vmin; v<=vmax; ++v) {
    double sg = ((v+j2+m2)&1) ? -1.0 : 1.0;
    S += sg*dfact(j2+j3+m1-v)*dfact(j1-m1+v)
         /(dfact(v)*dfact(j3-j1+j2-v)*dfact(j3+m3-v)*dfact(v+j1-j2-m3));
  }
  return C*S;
}

static __device__ void q_r2c(int l, double qr[7][7], double qi[7][7]){
  for (int a=0;a<7;++a) for (int b=0;b<7;++b) { qr[a][b]=0.0; qi[a][b]=0.0; }
  const double s = 1.0/sqrt(2.0);
  for (int m=-l; m<0; ++m) { qr[l+m][l-m] = s; qi[l+m][l+m] = -s; }
  qr[l][l] = 1.0;
  for (int m=1; m<=l; ++m) {
    double sg = (m&1) ? -1.0 : 1.0;
    qr[l+m][l+m] = sg*s; qi[l+m][l-m] = sg*s;
  }
  int ph = l & 3; // (-1j)^l
  double pr = (ph==0)?1.0:((ph==2)?-1.0:0.0);
  double pi = (ph==1)?-1.0:((ph==3)?1.0:0.0);
  for (int a=0;a<7;++a) for (int b=0;b<7;++b) {
    double r=qr[a][b], i=qi[a][b];
    qr[a][b] = r*pr - i*pi; qi[a][b] = r*pi + i*pr;
  }
}

__global__ void cg_kernel(float* __restrict__ cgtab){
  int p = threadIdx.x;
  if (p >= NPATH) return;
  int l1=cL1[p], l2=cL2[p], l3=cLO[p];
  int d1=2*l1+1, d2=2*l2+1, d3=2*l3+1;
  double C[7][7][7];
  for (int i=0;i<7;++i) for (int k=0;k<7;++k) for (int n=0;n<7;++n) C[i][k][n]=0.0;
  #pragma clang loop unroll(disable)
  for (int m1=-l1;m1<=l1;++m1)
    for (int m2=-l2;m2<=l2;++m2) {
      int m3=m1+m2;
      if (m3>=-l3 && m3<=l3) C[l1+m1][l2+m2][l3+m3] = su2_cg(l1,m1,l2,m2,l3,m3);
    }
  double Q1r[7][7],Q1i[7][7],Q2r[7][7],Q2i[7][7],Q3r[7][7],Q3i[7][7];
  q_r2c(l1,Q1r,Q1i); q_r2c(l2,Q2r,Q2i); q_r2c(l3,Q3r,Q3i);
  double outv[160];
  double nrm = 0.0;
  #pragma clang loop unroll(disable)
  for (int j=0;j<d1;++j) for (int l=0;l<d2;++l) for (int m=0;m<d3;++m) {
    double ar=0.0;
    for (int i=0;i<d1;++i) for (int k=0;k<d2;++k) {
      double q12r = Q1r[i][j]*Q2r[k][l] - Q1i[i][j]*Q2i[k][l];
      double q12i = Q1r[i][j]*Q2i[k][l] + Q1i[i][j]*Q2r[k][l];
      for (int n=0;n<d3;++n) {
        double c = C[i][k][n];
        if (c == 0.0) continue;
        double qr = Q3r[n][m], qi = -Q3i[n][m];  // conj(Q3[n][m])
        ar += c*(q12r*qr - q12i*qi);
      }
    }
    outv[(j*d2+l)*d3+m] = ar;
    nrm += ar*ar;
  }
  double fan = (l3==0) ? 384.0 : 640.0;                  // element-norm fan-in
  double scale = sqrt((2.0*l3+1.0)/fan) / sqrt(nrm);     // path weight / ||w3j||
  float* dst = cgtab + cCGOFF[p];
  int tot = d1*d2*d3;
  #pragma clang loop unroll(disable)
  for (int q=0;q<tot;++q) dst[q] = (float)(outv[q]*scale);
}

// ---------------- weights -> bf16 WMMA B-fragment layout ----------------
__global__ __launch_bounds__(256) void wfrag_kernel(const float* __restrict__ W, __bf16* __restrict__ wf){
  int tid = blockIdx.x*256 + threadIdx.x;
  if (tid >= N_WFRAG*512) return;
  int g = tid >> 9, r = tid & 511;
  int lane = r >> 4, e = r & 15;
  int io, local;
  if (g < 96)       { io=0; local=g;      }
  else if (g < 256) { io=1; local=g-96;   }
  else              { io=2; local=g-256;  }
  int kc = local >> 3, nt = local & 7;
  int Kl = ((e < 8) ? e : 8 + e) + ((lane >> 4) << 3);      // B 32x16 bf16: K = e%8 + 8*(lane>=16) + 16*(e>=8)
  int n  = lane & 15;
  int Ucat = kc*32 + Kl;
  int slot = Ucat >> 7, u = Ucat & 127;
  int p = cIOP[io][slot];
  wf[tid] = (__bf16)W[p*16384 + u*128 + nt*16 + n];
}

// ---------------- main fused TP kernel ----------------
__global__ __launch_bounds__(256) void tp_main(const float* __restrict__ x,
                                               const float* __restrict__ y,
                                               const float* __restrict__ cgtab,
                                               const __bf16* __restrict__ wfrag,
                                               float* __restrict__ out){
  extern __shared__ unsigned char smem[];
  float*  Gsh = (float*)smem;                       // 16 nodes x 145 floats
  __bf16* Tsh = (__bf16*)(smem + T_BYTE_OFF);       // 172 A-fragments x 512 bf16
  const int t = threadIdx.x;
  const int zbase = blockIdx.x * 16;

  // ---- phase 0: G_{z,i,k} = sum_j cg_{ijk} y_{z,j}  (one thread per (node, path)) ----
  if (t < 16*NPATH) {
    const int z = t & 15, p = t >> 4;                // p in [0,13)
    const int zg = zbase + z;
    const int l1=cL1[p], l2=cL2[p], lo=cLO[p];
    const int d1=2*l1+1, d2=2*l2+1, dk=2*lo+1;
    const float* cg = cgtab + cCGOFF[p];
    const float* yb = y + (size_t)zg*16 + cYOFF[l2];
    float* Gb = Gsh + z*145 + cGOFF[p];
    for (int i=0;i<d1;++i)
      for (int k=0;k<dk;++k) {
        float s = 0.f;
        for (int j=0;j<d2;++j) s += cg[(i*d2+j)*dk + k] * yb[j];
        Gb[i*dk+k] = s;
      }
  }
  __syncthreads();

  // ---- phase 1: T_{z,u,k} = sum_i x_{z,u,i} G_{z,i,k}, stored bf16 in A-fragment layout ----
  {
    const int z = t >> 4, ul = t & 15;
    const int zg = zbase + z;
    const float* xg = x + (size_t)zg*1152;
    const float* G  = Gsh + z*145;
    #pragma clang loop unroll(disable)
    for (int r=0;r<8;++r) {
      const int u = ul + 16*r;
      float xv[9];
      xv[0] = xg[u];
      #pragma unroll
      for (int i=0;i<3;++i) xv[1+i] = xg[128 + u*3 + i];
      #pragma unroll
      for (int i=0;i<5;++i) xv[4+i] = xg[512 + u*5 + i];
      #pragma unroll
      for (int p=0;p<NPATH;++p) {                    // fully unrolled: all metadata constant-folds
        const int l1=cL1[p], lo=cLO[p], io=lo;
        const int d1=2*l1+1, dk=2*lo+1;
        const float* Gb = G + cGOFF[p];
        float acc[5];
        #pragma unroll
        for (int k=0;k<dk;++k) acc[k]=0.f;
        #pragma unroll
        for (int i=0;i<d1;++i) {
          const float xi = xv[cXL[l1] + i];
          #pragma unroll
          for (int k=0;k<dk;++k) acc[k] += xi * Gb[i*dk+k];
        }
        // A-fragment addressing (16x32 bf16): lane = M + 16*((K%16)>=8), elem = K%8 + 8*(K>=16)
        const int Ucat = cSLOT[p]*128 + u;
        const int kc = Ucat >> 5, Kl = Ucat & 31;
        const int sub = ((((Kl & 15) < 8) ? z : z + 16) << 4) + (Kl & 7) + ((Kl >> 4) << 3);
        #pragma unroll
        for (int k=0;k<dk;++k) {
          const int f = (io==0) ? kc : ((io==1) ? (12 + k*20 + kc) : (72 + k*20 + kc));
          Tsh[f*512 + sub] = (__bf16)acc[k];
        }
      }
    }
  }
  __syncthreads();

  // ---- phase 2: per-wave WMMA over the 72 output 16x16 tiles (exactly 9 per wave) ----
  {
    const int wave = t >> 5, lane = t & 31;
    #pragma clang loop unroll(disable)
    for (int it = 0; it < 9; ++it) {
      const int tile = wave + it*8;
      int io, k, nt;
      if (tile < 8)       { io=0; k=0;            nt=tile;        }
      else if (tile < 32) { io=1; k=(tile-8)>>3;  nt=(tile-8)&7;  }
      else                { io=2; k=(tile-32)>>3; nt=(tile-32)&7; }
      const int kcN   = (io==0) ? 12 : 20;
      const int fbase = (io==0) ? 0 : ((io==1) ? (12 + k*20) : (72 + k*20));
      const int gbase = cWFB[io] + nt;
      v8f acc = {0.f,0.f,0.f,0.f,0.f,0.f,0.f,0.f};
      for (int kc=0; kc<kcN; ++kc) {
        const v16bf a = *(const v16bf*)(Tsh + (size_t)(fbase + kc)*512 + lane*16);
        const v16bf b = *(const v16bf*)(wfrag + ((size_t)(gbase + kc*8)*32 + lane)*16);
        acc = __builtin_amdgcn_wmma_f32_16x16x32_bf16(false, a, false, b, (short)0, acc, false, false);
      }
      const int n = lane & 15, mh = lane >> 4;
      const int dk = 2*io + 1;                         // lo == io here
      const int col0 = cOUT[io] + (nt*16 + n)*dk + k;
      #pragma unroll
      for (int r=0;r<8;++r) {
        const int zg = zbase + r + 8*mh;               // D: VGPR r -> M row r (+8 for upper lanes)
        out[(size_t)zg*1152 + col0] = acc[r];
      }
    }
  }
}

extern "C" void kernel_launch(void* const* d_in, const int* in_sizes, int n_in,
                              void* d_out, int out_size, void* d_ws, size_t ws_size,
                              hipStream_t stream) {
  const float* x = (const float*)d_in[0];
  const float* y = (const float*)d_in[1];
  const float* W = (const float*)d_in[2];
  float* out = (float*)d_out;

  float*  cgtab = (float*)d_ws;
  __bf16* wf    = (__bf16*)((char*)d_ws + WF_BYTE_OFF);

  (void)hipFuncSetAttribute((const void*)tp_main,
                            hipFuncAttributeMaxDynamicSharedMemorySize, SMEM_BYTES);

  cg_kernel<<<1, 32, 0, stream>>>(cgtab);
  wfrag_kernel<<<(N_WFRAG*512 + 255)/256, 256, 0, stream>>>(W, wf);
  tp_main<<<65536/16, 256, SMEM_BYTES, stream>>>(x, y, cgtab, wf, out);
}
